// RGIN_86114094285427
// MI455X (gfx1250) — compile-verified, matrix operand order checked by reading, not run on hardware
//
#include <hip/hip_runtime.h>
#include <hip/hip_bf16.h>

#define NN   50000
#define EE   600000
#define FDIM 128
#define RREL 8
#define BBAS 4

typedef __attribute__((ext_vector_type(16))) __bf16 v16bf;
typedef __attribute__((ext_vector_type(2)))  __bf16 v2bf;
typedef __attribute__((ext_vector_type(8)))  float  v8f;

// Scalar converter (round-to-nearest-even) for the tiny prep kernel.
static __device__ __forceinline__ __bf16 f2bf(float f) {
  union { float f; unsigned u; } v; v.f = f;
  unsigned r = v.u + 0x7FFFu + ((v.u >> 16) & 1u);
  unsigned short h = (unsigned short)(r >> 16);
  return __builtin_bit_cast(__bf16, h);
}

// Packed pair convert: one native v_cvt_pk_bf16_f32 per pair.
static __device__ __forceinline__ v2bf cvt2(float lo, float hi) {
#if __has_builtin(__builtin_amdgcn_cvt_pk_bf16_f32)
  return __builtin_amdgcn_cvt_pk_bf16_f32(lo, hi);
#else
  unsigned out;
  asm("v_cvt_pk_bf16_f32 %0, %1, %2" : "=v"(out) : "v"(lo), "v"(hi));
  return __builtin_bit_cast(v2bf, out);
#endif
}

union ABfrag { v16bf v; v2bf p[8]; };

// A fragment (no scale): 16x32 bf16 tile from row-major f32 source; `row`
// points at this lane's row (M = lane&15). ISA K mapping: VGPR v -> K = kk +
// (v>>2)*16 + half*8 + (v&3)*2 (+bit0). VGPRs 0-3 cover 8 contiguous floats
// at kk+half*8, VGPRs 4-7 cover 8 contiguous floats at kk+16+half*8 -> 4x b128.
static __device__ __forceinline__ v16bf load_a_row(const float* row, int kk, int half) {
  const float* p0 = row + kk + (half << 3);
  const float4 fA0 = *(const float4*)(p0);
  const float4 fA1 = *(const float4*)(p0 + 4);
  const float4 fB0 = *(const float4*)(p0 + 16);
  const float4 fB1 = *(const float4*)(p0 + 20);
  ABfrag u;
  u.p[0] = cvt2(fA0.x, fA0.y);
  u.p[1] = cvt2(fA0.z, fA0.w);
  u.p[2] = cvt2(fA1.x, fA1.y);
  u.p[3] = cvt2(fA1.z, fA1.w);
  u.p[4] = cvt2(fB0.x, fB0.y);
  u.p[5] = cvt2(fB0.z, fB0.w);
  u.p[6] = cvt2(fB1.x, fB1.y);
  u.p[7] = cvt2(fB1.z, fB1.w);
  return u.v;
}

__global__ __launch_bounds__(256) void rgin_zero_kernel(float4* p, long n4) {
  long i = (long)blockIdx.x * blockDim.x + threadIdx.x;
  if (i < n4) p[i] = make_float4(0.f, 0.f, 0.f, 0.f);
}

// Build bf16 B-matrices in WMMA fragment-swizzled layout:
// element e of a 128x128 matrix: j = e&15, lane = (e>>4)&31, t = e>>9
// (t = wave*4 + kki). Source: n = (t>>2)*16 + (lane&15);
// k = (t&3)*32 + (v>>2)*16 + (lane>>4)*8 + (v&3)*2 + (j&1), v = j>>1.
// mat 0..7: wc[r] = sum_b comp[r,b]*weight[b]; 8: root; 9: w1*(1+eps); 10: w2.
__global__ __launch_bounds__(256) void rgin_prep_kernel(
    const float* __restrict__ comp, const float* __restrict__ weight,
    const float* __restrict__ root, const float* __restrict__ w1,
    const float* __restrict__ w2, const float* __restrict__ epsp,
    __bf16* __restrict__ bmats) {
  int d = blockIdx.x * 256 + threadIdx.x;
  if (d >= 11 * 16384) return;
  int mat = d >> 14;
  int e = d & 16383;
  int j = e & 15;
  int lane = (e >> 4) & 31;
  int t = e >> 9;
  int v = j >> 1;
  int n = ((t >> 2) << 4) + (lane & 15);
  int k = ((t & 3) << 5) + ((v >> 2) << 4) + ((lane >> 4) << 3) + ((v & 3) << 1) + (j & 1);
  int src = k * FDIM + n;
  float val;
  if (mat < 8) {
    float acc = 0.f;
#pragma unroll
    for (int b = 0; b < BBAS; ++b)
      acc += comp[mat * BBAS + b] * weight[(size_t)b * 16384 + src];
    val = acc;
  } else if (mat == 8) {
    val = root[src];
  } else if (mat == 9) {
    val = w1[src] * (1.0f + epsp[0]);
  } else {
    val = w2[src];
  }
  bmats[d] = f2bf(val);
}

// One wave (32 lanes) per edge, 4 floats per lane -> hardware f32 atomics.
__global__ __launch_bounds__(256) void rgin_scatter_kernel(const float* __restrict__ x,
                                                           const int* __restrict__ eidx,
                                                           const int* __restrict__ etype,
                                                           float* __restrict__ sums,
                                                           float* __restrict__ cnt) {
  long tid = (long)blockIdx.x * 256 + threadIdx.x;
  long e = tid >> 5;
  int lane = (int)(tid & 31);
  if (e >= EE) return;
  int s = eidx[e];
  int d = eidx[EE + e];
  int t = etype[e];
  long seg = (long)d * RREL + t;
  const float4 v = *(const float4*)(x + (size_t)s * FDIM + lane * 4);
  float* o = sums + seg * FDIM + lane * 4;
  unsafeAtomicAdd(o + 0, v.x);
  unsafeAtomicAdd(o + 1, v.y);
  unsafeAtomicAdd(o + 2, v.z);
  unsafeAtomicAdd(o + 3, v.w);
  if (lane == 0) unsafeAtomicAdd(cnt + seg, 1.0f);
}

// Block = 16 nodes, 8 waves; wave w owns output columns [16w, 16w+16).
// bmats: swizzled bf16 [11][16384]: 0..7 = wc[r], 8 = root, 9 = w1*(1+eps), 10 = w2.
__global__ __launch_bounds__(256) void rgin_node_kernel(
    const float* __restrict__ xin, const float* __restrict__ sums,
    const float* __restrict__ cnt, const __bf16* __restrict__ bmats,
    const float* __restrict__ bias, const float* __restrict__ b1,
    const float* __restrict__ ln_g, const float* __restrict__ ln_b,
    const float* __restrict__ b2, float* __restrict__ xout, int do_relu) {
  __shared__ float lds_x[16 * FDIM];
  __shared__ float lds_h[16 * FDIM];
  __shared__ float lds_rc[16 * RREL];
  __shared__ float lds_mu[16];
  __shared__ float lds_rs[16];

  const int tid = threadIdx.x;
  const int wave = tid >> 5;
  const int lane = tid & 31;
  const int half = lane >> 4;
  const int am = lane & 15;
  const int node0 = blockIdx.x * 16;
  const int ncol = wave * 16;

  // Stage x tile and per-(node, rel) reciprocal counts.
  for (int j = tid; j < 16 * FDIM; j += 256)
    lds_x[j] = xin[(size_t)(node0 + (j >> 7)) * FDIM + (j & 127)];
  if (tid < 16 * RREL) {
    float c = cnt[(size_t)(node0 + (tid >> 3)) * RREL + (tid & 7)];
    lds_rc[tid] = 1.0f / fmaxf(c, 1.0f);
  }
  __syncthreads();

  // Per-wave B fragment index base: fragment f(kki) at (wave*4+kki)*32 + lane.
  const v16bf* bfrag = (const v16bf*)bmats;  // [11*32*32] v16bf
  const int fbase = wave * 4 * 32 + lane;
  const int crow0 = half << 3;  // C-layout: element d sits at row d + 8*half

  v8f acc1 = {0.f, 0.f, 0.f, 0.f, 0.f, 0.f, 0.f, 0.f};

  // ---- relation GEMM: out += (1/cnt[n,r]) * (sum[n, r*128+k] @ wc[r][k][o]) ----
  // Mean division hoisted out of the K-loop: accumulate each relation into a
  // temp C tile with unscaled A, then FMA-fold rc per C row (f32, post-WMMA).
  for (int r = 0; r < RREL; ++r) {
    const float* arow = sums + ((size_t)(node0 + am) * RREL + r) * FDIM;
    const v16bf* bm = bfrag + (size_t)r * 1024 + fbase;
    __builtin_prefetch(bfrag + (size_t)((r + 1) & 7) * 1024 + fbase, 0, 0);
    v8f tacc = {0.f, 0.f, 0.f, 0.f, 0.f, 0.f, 0.f, 0.f};
#pragma unroll
    for (int kki = 0; kki < 4; ++kki) {
      v16bf a = load_a_row(arow, kki << 5, half);
      v16bf b = bm[kki * 32];
      tacc = __builtin_amdgcn_wmma_f32_16x16x32_bf16(false, a, false, b, (short)0, tacc, false, false);
    }
#pragma unroll
    for (int d = 0; d < 8; ++d)
      acc1[d] = fmaf(tacc[d], lds_rc[(crow0 + d) * RREL + r], acc1[d]);
  }

  // ---- root GEMM and w1 GEMM share the same A fragments (w1 pre-scaled) ----
  v8f acc2 = {0.f, 0.f, 0.f, 0.f, 0.f, 0.f, 0.f, 0.f};
  {
    const v16bf* bR = bfrag + (size_t)8 * 1024 + fbase;
    const v16bf* bW1 = bfrag + (size_t)9 * 1024 + fbase;
#pragma unroll
    for (int kki = 0; kki < 4; ++kki) {
      v16bf a = load_a_row(lds_x + am * FDIM, kki << 5, half);
      v16bf br = bR[kki * 32];
      acc1 = __builtin_amdgcn_wmma_f32_16x16x32_bf16(false, a, false, br, (short)0, acc1, false, false);
      v16bf bw = bW1[kki * 32];
      acc2 = __builtin_amdgcn_wmma_f32_16x16x32_bf16(false, a, false, bw, (short)0, acc2, false, false);
    }
  }
  {
    float bc = bias[ncol + am];
#pragma unroll
    for (int d = 0; d < 8; ++d) acc1[d] += bc;
  }

  // h1 tile -> LDS (C/D layout: row M = d + 8*half, col = lane&15)
  {
    float bc = b1[ncol + am];
#pragma unroll
    for (int d = 0; d < 8; ++d)
      lds_h[(crow0 + d) * FDIM + ncol + am] = acc2[d] + bc;
  }
  __syncthreads();

  // ---- LayerNorm + ReLU over full rows in LDS ----
  if (tid < 16) {
    float mu = 0.f;
    for (int j = 0; j < FDIM; ++j) mu += lds_h[tid * FDIM + j];
    mu *= (1.0f / FDIM);
    float var = 0.f;
    for (int j = 0; j < FDIM; ++j) {
      float dd = lds_h[tid * FDIM + j] - mu;
      var += dd * dd;
    }
    var *= (1.0f / FDIM);
    lds_mu[tid] = mu;
    lds_rs[tid] = rsqrtf(var + 1e-5f);
  }
  __syncthreads();
  for (int j = tid; j < 16 * FDIM; j += 256) {
    int row = j >> 7, c = j & 127;
    float v = (lds_h[j] - lds_mu[row]) * lds_rs[row] * ln_g[c] + ln_b[c];
    lds_h[j] = fmaxf(v, 0.0f);
  }
  __syncthreads();

  // ---- h2 = relu(ln(h1)) @ w2 + b2 ; final = h2 + acc1 (+ outer relu) ----
  {
    v8f acc3 = {0.f, 0.f, 0.f, 0.f, 0.f, 0.f, 0.f, 0.f};
    const v16bf* bW2 = bfrag + (size_t)10 * 1024 + fbase;
#pragma unroll
    for (int kki = 0; kki < 4; ++kki) {
      v16bf a = load_a_row(lds_h + am * FDIM, kki << 5, half);
      v16bf b = bW2[kki * 32];
      acc3 = __builtin_amdgcn_wmma_f32_16x16x32_bf16(false, a, false, b, (short)0, acc3, false, false);
    }
    float bc = b2[ncol + am];
#pragma unroll
    for (int d = 0; d < 8; ++d) {
      float res = acc3[d] + bc + acc1[d];
      if (do_relu) res = fmaxf(res, 0.0f);
      xout[(size_t)(node0 + crow0 + d) * FDIM + ncol + am] = res;
    }
  }
}

extern "C" void kernel_launch(void* const* d_in, const int* in_sizes, int n_in,
                              void* d_out, int out_size, void* d_ws, size_t ws_size,
                              hipStream_t stream) {
  (void)in_sizes; (void)n_in; (void)out_size; (void)ws_size;
  const float* x     = (const float*)d_in[0];
  const int*   eidx  = (const int*)d_in[1];
  const int*   etype = (const int*)d_in[2];

  // Workspace carve-up (all chunks 32B-aligned).
  char* w = (char*)d_ws;
  float*  sums  = (float*)w;  w += (size_t)NN * RREL * FDIM * sizeof(float);  // 204.8 MB
  float*  cnt   = (float*)w;  w += (size_t)NN * RREL * sizeof(float);         // 1.6 MB
  __bf16* bmats = (__bf16*)w; w += (size_t)11 * 16384 * sizeof(__bf16);       // 352 KB
  float*  u0    = (float*)w;  w += (size_t)NN * FDIM * sizeof(float);         // 25.6 MB
  float*  u1    = (float*)w;                                                  // 25.6 MB

  const float* lin[3]  = {x, u0, u1};
  float*       lout[3] = {u0, u1, (float*)d_out};

  const long zero_n4   = (long)NN * RREL * (FDIM + 1) / 4;  // sums + cnt (contiguous)
  const int  zero_grid = (int)((zero_n4 + 255) / 256);
  const int  prep_grid = (11 * 16384 + 255) / 256;
  const long scat_thr  = (long)EE * 32;
  const int  scat_grid = (int)((scat_thr + 255) / 256);
  const int  node_grid = NN / 16;  // 3125

  for (int l = 0; l < 3; ++l) {
    const int base = 3 + 11 * l;
    const float* p_weight = (const float*)d_in[base + 0];
    const float* p_comp   = (const float*)d_in[base + 1];
    const float* p_root   = (const float*)d_in[base + 2];
    const float* p_bias   = (const float*)d_in[base + 3];
    const float* p_eps    = (const float*)d_in[base + 4];
    const float* p_w1     = (const float*)d_in[base + 5];
    const float* p_b1     = (const float*)d_in[base + 6];
    const float* p_ln_g   = (const float*)d_in[base + 7];
    const float* p_ln_b   = (const float*)d_in[base + 8];
    const float* p_w2     = (const float*)d_in[base + 9];
    const float* p_b2     = (const float*)d_in[base + 10];

    rgin_zero_kernel<<<zero_grid, 256, 0, stream>>>((float4*)sums, zero_n4);
    rgin_prep_kernel<<<prep_grid, 256, 0, stream>>>(p_comp, p_weight, p_root, p_w1,
                                                    p_w2, p_eps, bmats);
    rgin_scatter_kernel<<<scat_grid, 256, 0, stream>>>(lin[l], eidx, etype, sums, cnt);
    rgin_node_kernel<<<node_grid, 256, 0, stream>>>(
        lin[l], sums, cnt, bmats, p_bias, p_b1, p_ln_g, p_ln_b, p_b2,
        lout[l], (l < 2) ? 1 : 0);
  }
}
